// CarafeUpsample_70471823393274
// MI455X (gfx1250) — compile-verified, matrix operand order checked by reading, not run on hardware
//
#include <hip/hip_runtime.h>
#include <hip/hip_bf16.h>

typedef __attribute__((ext_vector_type(16))) _Float16 v16h;
typedef __attribute__((ext_vector_type(8)))  _Float16 v8h;
typedef __attribute__((ext_vector_type(8)))  float    v8f;

// Problem constants (from reference): B=2, C=256, COMP=64, H=W=64, S=2, K=5
#define NB    2
#define NC    256
#define NCOMP 64
#define NH    64
#define NW    64
#define NHW   4096          // 64*64
#define NPIX  8192          // B*H*W
#define KPC   100           // S^2*K^2 predicted-kernel channels
#define KPCP  112           // padded to 7 WMMA M-tiles
#define HS    128
#define K2    25

// Build a 16-half WMMA fragment from its two contiguous 8-half runs.
// CDNA5 16-bit fragment layout (CDNA5 ISA): lane l (hilo = l>=16 ? 8 : 0)
// holds halves k = kb+hilo+0..7 (VGPR0-3) and k = kb+16+hilo+0..7 (VGPR4-7).
__device__ __forceinline__ v16h frag_from(const _Float16* base, int kb, int hilo) {
    const v8h lo = *(const v8h*)(base + kb + hilo);
    const v8h hi = *(const v8h*)(base + kb + 16 + hilo);
    return __builtin_shufflevector(lo, hi, 0,1,2,3,4,5,6,7,8,9,10,11,12,13,14,15);
}

// ---------------------------------------------------------------------------
// Prep 0: w_enc (64x256 f32, already k-contiguous) -> f16
// ---------------------------------------------------------------------------
__global__ __launch_bounds__(256) void cvt_wenc(const float* __restrict__ w_enc,
                                                _Float16* __restrict__ w16)
{
    const int i = blockIdx.x * blockDim.x + threadIdx.x;   // 0..16383
    if (i < NCOMP * NC) w16[i] = (_Float16)w_enc[i];
}

// ---------------------------------------------------------------------------
// Prep 1: transpose x [b][c][hw] f32 -> x16T [b][hw][c] f16 (c contiguous).
// Thread = one pixel; reads coalesced across lanes, writes 16B per step.
// ---------------------------------------------------------------------------
__global__ __launch_bounds__(256) void transpose_x(const float* __restrict__ x,
                                                   _Float16* __restrict__ x16T)
{
    const int n = blockIdx.x * blockDim.x + threadIdx.x;   // 0..8191 (b*4096+hw)
    if (n >= NPIX) return;
    const int b = n >> 12, hw = n & 4095;
    const float* src = x + (size_t)b * NC * NHW + hw;      // stride NHW over c
    _Float16* dst = x16T + (size_t)n * NC;
#pragma unroll 4
    for (int c0 = 0; c0 < NC; c0 += 8) {
        v8h v;
#pragma unroll
        for (int j = 0; j < 8; ++j) v[j] = (_Float16)src[(size_t)(c0 + j) * NHW];
        *(v8h*)(dst + c0) = v;
    }
}

// ---------------------------------------------------------------------------
// Prep 2: repack w_kp (100,64,3,3) f32 -> wkp16 [o=112][t=9][c=64] f16,
// zero-padded rows for o>=100, so GEMM-K = t*64+c is contiguous.
// ---------------------------------------------------------------------------
__global__ __launch_bounds__(256) void cvt_wkp(const float* __restrict__ w_kp,
                                               _Float16* __restrict__ wkp16)
{
    const int i = blockIdx.x * blockDim.x + threadIdx.x;   // 0..64511
    if (i >= KPCP * 9 * NCOMP) return;
    const int o = i / (9 * NCOMP);
    const int r = i % (9 * NCOMP);
    const int t = r >> 6, c = r & 63;
    wkp16[i] = (o < KPC) ? (_Float16)w_kp[(o * NCOMP + c) * 9 + t] : (_Float16)0.0f;
}

// ---------------------------------------------------------------------------
// Stage 1: content encoder 1x1 conv as WMMA GEMM.
//   M=64 (4 tiles) x N=8192 pixels (512 tiles), K=256 (8 chunks of 32).
//   All fragment loads are aligned b128; D stored as one 16B f16 run per lane
//   into xc16T [pixel][c] (rows M=r+hilo are contiguous in that layout).
// ---------------------------------------------------------------------------
__global__ __launch_bounds__(256) void enc_gemm(const _Float16* __restrict__ x16T,
                                                const _Float16* __restrict__ w16,
                                                const float* __restrict__ b_enc,
                                                _Float16* __restrict__ xc16T)
{
    const int lane = threadIdx.x & 31;
    const int wave = (blockIdx.x * blockDim.x + threadIdx.x) >> 5; // 0..2047
    const int mt = wave & 3;        // 4 M-tiles
    const int nt = wave >> 2;       // 512 N-tiles
    const int lm = lane & 15;
    const int hilo = (lane >> 4) << 3;

    const int n = nt * 16 + lm;                 // pixel column (b*4096+hw)
    const int m = mt * 16 + lm;                 // output channel row
    const _Float16* arow = w16 + m * NC;
    const _Float16* brow = x16T + (size_t)n * NC;

    v8f acc = {};
#pragma unroll
    for (int kb = 0; kb < NC; kb += 32) {
        const v16h a  = frag_from(arow, kb, hilo);
        const v16h bf = frag_from(brow, kb, hilo);
        acc = __builtin_amdgcn_wmma_f32_16x16x32_f16(false, a, false, bf,
                                                     (short)0, acc, false, false);
    }
    const int mo0 = mt * 16 + hilo;             // D rows M = r + hilo, contiguous
    v8h st;
#pragma unroll
    for (int r = 0; r < 8; ++r) st[r] = (_Float16)(acc[r] + b_enc[mo0 + r]);
    *(v8h*)(xc16T + (size_t)n * NCOMP + mo0) = st;
}

// ---------------------------------------------------------------------------
// Stage 2: 3x3 kernel-predictor conv as implicit WMMA GEMM.
//   M=112 (7 tiles), N=8192 (512 tiles), K=576 (18 chunks of 32; each chunk
//   lives inside one 3x3 tap since k = t*64 + c). B comes from xc16T at the
//   tap-shifted pixel; the bounds test is uniform per chunk.
// ---------------------------------------------------------------------------
__global__ __launch_bounds__(256) void kp_gemm(const _Float16* __restrict__ xc16T,
                                               const _Float16* __restrict__ wkp16,
                                               const float* __restrict__ b_kp,
                                               float* __restrict__ kpredT)
{
    const int lane = threadIdx.x & 31;
    const int wave = (blockIdx.x * blockDim.x + threadIdx.x) >> 5; // 0..3583
    const int mt = wave % 7;
    const int nt = wave / 7;
    const int lm = lane & 15;
    const int hilo = (lane >> 4) << 3;

    const int n    = nt * 16 + lm;
    const int bimg = n >> 12;
    const int hw   = n & 4095;
    const int h    = hw >> 6;
    const int w    = hw & 63;
    const int o    = mt * 16 + lm;
    const _Float16* arow = wkp16 + (size_t)o * (9 * NCOMP);

    v8f acc = {};
#pragma unroll
    for (int kc = 0; kc < 18; ++kc) {
        const int kb    = kc * 32;
        const int t     = kb >> 6;       // tap 0..8 (uniform per chunk)
        const int cbase = kb & 63;       // 0 or 32
        const int dy = t / 3 - 1, dx = t % 3 - 1;
        const int hh = h + dy, wq = w + dx;
        const bool inb = (hh >= 0) & (hh < NH) & (wq >= 0) & (wq < NW);
        const _Float16* brow =
            xc16T + (size_t)((bimg << 12) + hh * NW + wq) * NCOMP;

        const v16h a = frag_from(arow, kb, hilo);
        v16h bf;
        if (inb) {
            bf = frag_from(brow, cbase, hilo);
            if (kc + 2 < 18)   // prefetch B for the chunk after next (gfx1250 path)
                __builtin_prefetch(brow + NCOMP, 0, 0);
        } else {
            bf = (v16h)(_Float16)0.0f;
        }
        acc = __builtin_amdgcn_wmma_f32_16x16x32_f16(false, a, false, bf,
                                                     (short)0, acc, false, false);
    }
    // Store pixel-major kpredT [n][112]; rows M=r+hilo contiguous -> v8f store.
    const int oo0 = mt * 16 + hilo;
    v8f st;
#pragma unroll
    for (int r = 0; r < 8; ++r) {
        const int oo = oo0 + r;
        st[r] = acc[r] + ((oo < KPC) ? b_kp[oo] : 0.0f);
    }
    *(v8f*)(kpredT + (size_t)n * KPCP + oo0) = st;
}

// ---------------------------------------------------------------------------
// Stage 3: pixel-shuffle + softmax over the 25 reassembly weights.
//   channel = k2*4 + p*2 + q at low-res pixel (h,w); kpredT is pixel-major
//   so the 25 reads are a strided walk within one 448B row (L2-resident).
// ---------------------------------------------------------------------------
__global__ __launch_bounds__(256) void softmax_shuffle(const float* __restrict__ kpredT,
                                                       float* __restrict__ wgt)
{
    const int idx = blockIdx.x * blockDim.x + threadIdx.x;   // 0..32767
    if (idx >= NB * HS * HS) return;
    const int b   = idx >> 14;
    const int rem = idx & 16383;
    const int hs = rem >> 7, ws = rem & 127;
    const int h = hs >> 1, p = hs & 1;
    const int w = ws >> 1, q = ws & 1;
    const int off = p * 2 + q;
    const float* row = kpredT + (size_t)((b << 12) + h * NW + w) * KPCP + off;

    float v[K2];
    float mx = -1e30f;
#pragma unroll
    for (int k = 0; k < K2; ++k) {
        v[k] = row[k * 4];
        mx = fmaxf(mx, v[k]);
    }
    float s = 0.0f;
#pragma unroll
    for (int k = 0; k < K2; ++k) { v[k] = __expf(v[k] - mx); s += v[k]; }
    const float inv = 1.0f / s;
    float* wp = wgt + (size_t)idx * K2;
#pragma unroll
    for (int k = 0; k < K2; ++k) wp[k] = v[k] * inv;
}

// ---------------------------------------------------------------------------
// Stage 4: CARAFE reassembly (memory-bound streaming, fp32 throughout).
//   Block = one (b,c,hs) row, thread = ws -> x reads coalesced across lanes,
//   weight reads are 25 contiguous floats per thread (192MB L2 absorbs reuse).
// ---------------------------------------------------------------------------
__global__ __launch_bounds__(128) void reassemble(const float* __restrict__ x,
                                                  const float* __restrict__ wgt,
                                                  float* __restrict__ out)
{
    const int ws  = threadIdx.x;             // 0..127
    const int bid = blockIdx.x;              // (b*256 + c)*128 + hs
    const int hs  = bid & 127;
    const int bc  = bid >> 7;                // b*256 + c
    const int b   = bc >> 8;
    const int h   = hs >> 1;
    const int w   = ws >> 1;

    const float* xp = x + (size_t)bc * NHW;
    const float* wp = wgt + ((size_t)(b * (HS * HS) + hs * HS + ws)) * K2;

    float acc = 0.0f;
#pragma unroll
    for (int k = 0; k < K2; ++k) {
        const int di = k / 5 - 2, dj = k % 5 - 2;
        const int hh = h + di, wq = w + dj;
        const float xv = (hh >= 0 && hh < NH && wq >= 0 && wq < NW)
                             ? xp[hh * NW + wq] : 0.0f;
        acc += xv * wp[k];
    }
    out[(size_t)bid * 128 + ws] = acc;
}

// ---------------------------------------------------------------------------
extern "C" void kernel_launch(void* const* d_in, const int* in_sizes, int n_in,
                              void* d_out, int out_size, void* d_ws, size_t ws_size,
                              hipStream_t stream)
{
    const float* x     = (const float*)d_in[0];
    const float* w_enc = (const float*)d_in[1];
    const float* b_enc = (const float*)d_in[2];
    const float* w_kp  = (const float*)d_in[3];
    const float* b_kp  = (const float*)d_in[4];
    float* out = (float*)d_out;

    // Workspace layout (all offsets 256B aligned):
    //   x16T   f16 [2][4096][256]   4,194,304 B
    //   w16    f16 [64][256]           32,768 B
    //   wkp16  f16 [112][9][64]       129,024 B
    //   xc16T  f16 [2][4096][64]    1,048,576 B
    //   kpredT f32 [2][4096][112]   3,670,016 B
    //   wgt    f32 [2][128][128][25] 3,276,800 B     total ~12.3 MB
    char* ws = (char*)d_ws;
    _Float16* x16T   = (_Float16*)(ws);
    _Float16* w16    = (_Float16*)(ws + 4194304u);
    _Float16* wkp16  = (_Float16*)(ws + 4194304u + 32768u);
    _Float16* xc16T  = (_Float16*)(ws + 4194304u + 32768u + 129024u);
    float*    kpredT = (float*)   (ws + 4194304u + 32768u + 129024u + 1048576u);
    float*    wgt    = (float*)   (ws + 4194304u + 32768u + 129024u + 1048576u + 3670016u);

    cvt_wenc<<<64, 256, 0, stream>>>(w_enc, w16);
    transpose_x<<<32, 256, 0, stream>>>(x, x16T);
    cvt_wkp<<<252, 256, 0, stream>>>(w_kp, wkp16);

    // 2048 waves = 4 x 512 tiles, 8 waves/block
    enc_gemm<<<256, 256, 0, stream>>>(x16T, w16, b_enc, xc16T);
    // 3584 waves = 7 x 512 tiles
    kp_gemm<<<448, 256, 0, stream>>>(xc16T, wkp16, b_kp, kpredT);
    // 32768 upsampled pixels
    softmax_shuffle<<<128, 256, 0, stream>>>(kpredT, wgt);
    // 65536 rows x 128 columns of output
    reassemble<<<NB * NC * HS, 128, 0, stream>>>(x, wgt, out);
}
